// NerfHash_37486474560114
// MI455X (gfx1250) — matrix operand order repaired and checked by reading, not compiled, not run on hardware
//
#include <hip/hip_runtime.h>

typedef _Float16 f16;
typedef __attribute__((ext_vector_type(16))) _Float16 v16h;
typedef __attribute__((ext_vector_type(8)))  float    v8f;

#define TBITS 19u
#define TSZ   (1u << TBITS)

__host__ __device__ constexpr int res_at(int l) {
  // floor(16 * 1.3819^l)
  constexpr int r[16] = {16, 22, 30, 42, 58, 80, 111, 153,
                         212, 294, 406, 561, 775, 1072, 1481, 2047};
  return r[l];
}

// ---------------- WMMA fragment helpers (ISA 7.12.2 layouts, wave32) ---------

// A-matrix 16x32 f16: lanes 0-15: M=lane, K = {0..7, 16..23}; lanes 16-31: K={8..15, 24..31}
__device__ __forceinline__ v16h ldsAfrag(const f16* A, int stride, int row0,
                                         int kchunk, int lane) {
  const int m  = lane & 15;
  const int kb = kchunk * 32 + ((lane & 16) ? 8 : 0);
  const f16* p = A + (row0 + m) * stride + kb;
  v16h a;
#pragma unroll
  for (int i = 0; i < 8; ++i) { a[i] = p[i]; a[8 + i] = p[16 + i]; }
  return a;
}

// B-matrix 32x16 f16 from K-contiguous (transposed) weights:
// lane holds column n0+(lane&15); lanes 0-15 -> K {0..7, 16..23}, lanes 16-31 -> K {8..15, 24..31}
__device__ __forceinline__ v16h ldsBfrag(const f16* Wt, int stride, int n0,
                                         int kchunk, int lane) {
  const int col = n0 + (lane & 15);
  const int kb  = kchunk * 32 + ((lane & 16) ? 8 : 0);
  const f16* p  = Wt + col * stride + kb;
  v16h b;
#pragma unroll
  for (int i = 0; i < 8; ++i) { b[i] = p[i]; b[8 + i] = p[16 + i]; }
  return b;
}

__device__ __forceinline__ v8f wmma16(v16h a, v16h b, v8f c) {
  return __builtin_amdgcn_wmma_f32_16x16x32_f16(false, a, false, b,
                                                (short)0, c, false, false);
}

// C/D f32 16x16: VGPR r -> row r (+8 for lanes 16-31), col = lane&15
__device__ __forceinline__ void storeCrelu(f16* H, int stride, int row0, int n0,
                                           int lane, v8f acc) {
  const int col = n0 + (lane & 15);
  const int rb  = (lane & 16) ? 8 : 0;
#pragma unroll
  for (int r = 0; r < 8; ++r)
    H[(row0 + rb + r) * stride + col] = (f16)fmaxf(acc[r], 0.0f);
}

// ---------------- hash-grid encode: 8 levels per thread ---------------------

template <int L0>
__device__ __forceinline__ void encode8(float x, float y, float z, f16* dst,
                                        const float* __restrict__ table) {
#pragma unroll
  for (int li = 0; li < 8; ++li) {
    const int l = L0 + li;
    const float res = (float)res_at(l);
    const float px = x * res, py = y * res, pz = z * res;
    const float fx = floorf(px), fy = floorf(py), fz = floorf(pz);
    const float wx = px - fx, wy = py - fy, wz = pz - fz;
    const unsigned ix = (unsigned)fx, iy = (unsigned)fy, iz = (unsigned)fz;
    const float2* tb = (const float2*)table + (size_t)l * TSZ;
    const unsigned s = (unsigned)res_at(l) + 1u;
    float f0 = 0.f, f1 = 0.f;
#pragma unroll
    for (int c = 0; c < 8; ++c) {
      const unsigned cx = ix + (c & 1);
      const unsigned cy = iy + ((c >> 1) & 1);
      const unsigned cz = iz + (c >> 2);
      unsigned idx;
      if (l < 5) {                      // dense level: (res+1)^3 <= T
        idx = cx + cy * s + cz * s * s;
      } else {                          // hashed level
        idx = (cx ^ (cy * 2654435761u) ^ (cz * 805459861u)) & (TSZ - 1u);
      }
      const float2 f = tb[idx];
      const float w = ((c & 1) ? wx : 1.f - wx) *
                      (((c >> 1) & 1) ? wy : 1.f - wy) *
                      ((c >> 2) ? wz : 1.f - wz);
      f0 += f.x * w;
      f1 += f.y * w;
    }
    dst[2 * l]     = (f16)f0;
    dst[2 * l + 1] = (f16)f1;
  }
}

// ---------------- fused NeRF kernel -----------------------------------------

__global__ __launch_bounds__(256) void nerf_fused(
    const float* __restrict__ xyz, const float* __restrict__ dir,
    const float* __restrict__ table,
    const float* __restrict__ ws1, const float* __restrict__ ws2,
    const float* __restrict__ wr1, const float* __restrict__ wr2,
    const float* __restrict__ wr3,
    float* __restrict__ rgb_out, float* __restrict__ sigma_out, int N) {
  // Activations (row-padded so every 16B A-fragment chunk is 16B aligned)
  __shared__ f16 sA[128 * 40];    // hash features, 32 cols        (10.0 KB)
  __shared__ f16 sH[128 * 72];    // 64-wide hidden (reused 3x)    (18.0 KB)
  __shared__ f16 sR[128 * 40];    // [SH dir enc | geo_feat]       (10.0 KB)
  // Weights, K-contiguous (transposed), f16
  __shared__ f16 sWs1[64 * 32];   // w_sigma1^T                    ( 4.0 KB)
  __shared__ f16 sWs2[32 * 64];   // w_sigma2^T, 17 -> 32 cols pad ( 4.0 KB)
  __shared__ f16 sWr1[64 * 32];   // w_rgb1^T                      ( 4.0 KB)
  __shared__ f16 sWr2[64 * 64];   // w_rgb2^T                      ( 8.0 KB)
  __shared__ f16 sWr3[16 * 64];   // w_rgb3^T,  3 -> 16 cols pad   ( 2.0 KB)

  const int tid  = threadIdx.x;
  const int base = blockIdx.x * 128;

  // ---- weights -> LDS (transposed, f16, zero padding), single pass ----
  for (int i = tid; i < 64 * 32; i += 256) {      // w_sigma1 (32x64)
    const int n = i >> 5, k = i & 31;
    sWs1[n * 32 + k] = (f16)ws1[k * 64 + n];
  }
  for (int i = tid; i < 32 * 64; i += 256) {      // w_sigma2 (64x17 -> 64x32)
    const int n = i >> 6, k = i & 63;
    sWs2[n * 64 + k] = (f16)((n < 17) ? ws2[k * 17 + n] : 0.0f);
  }
  for (int i = tid; i < 64 * 32; i += 256) {      // w_rgb1 (32x64)
    const int n = i >> 5, k = i & 31;
    sWr1[n * 32 + k] = (f16)wr1[k * 64 + n];
  }
  for (int i = tid; i < 64 * 64; i += 256) {      // w_rgb2 (64x64)
    const int n = i >> 6, k = i & 63;
    sWr2[n * 64 + k] = (f16)wr2[k * 64 + n];
  }
  for (int i = tid; i < 16 * 64; i += 256) {      // w_rgb3 (64x3 -> 64x16)
    const int n = i >> 6, k = i & 63;
    sWr3[n * 64 + k] = (f16)((n < 3) ? wr3[k * 3 + n] : 0.0f);
  }

  // ---- hash-grid encode: waves 0-3 do levels 0-7, waves 4-7 levels 8-15 ----
  {
    const int p  = tid & 127;
    const int gp = base + p;
    if (gp < N) {
      const float x = xyz[3 * gp + 0];
      const float y = xyz[3 * gp + 1];
      const float z = xyz[3 * gp + 2];
      if (tid < 128) encode8<0>(x, y, z, &sA[p * 40], table);
      else           encode8<8>(x, y, z, &sA[p * 40], table);
    }
  }

  // ---- SH degree-4 direction encoding -> sR cols 0..15 ----
  if (tid < 128) {
    const int q = base + tid;
    if (q < N) {
      const float dx = dir[3 * q + 0] * 2.f - 1.f;
      const float dy = dir[3 * q + 1] * 2.f - 1.f;
      const float dz = dir[3 * q + 2] * 2.f - 1.f;
      const float x2 = dx * dx, y2 = dy * dy, z2 = dz * dz;
      const float xy = dx * dy, yz = dy * dz, xz = dx * dz;
      f16* R = &sR[tid * 40];
      R[0]  = (f16)0.28209479177387814f;
      R[1]  = (f16)(-0.48860251190291987f * dy);
      R[2]  = (f16)( 0.48860251190291987f * dz);
      R[3]  = (f16)(-0.48860251190291987f * dx);
      R[4]  = (f16)( 1.0925484305920792f * xy);
      R[5]  = (f16)(-1.0925484305920792f * yz);
      R[6]  = (f16)( 0.94617469575756f * z2 - 0.31539156525252f);
      R[7]  = (f16)(-1.0925484305920792f * xz);
      R[8]  = (f16)( 0.5462742152960396f * (x2 - y2));
      R[9]  = (f16)( 0.5900435899266435f * dy * (-3.f * x2 + y2));
      R[10] = (f16)( 2.890611442640554f * xy * dz);
      R[11] = (f16)( 0.4570457994644657f * dy * (1.f - 5.f * z2));
      R[12] = (f16)( 0.3731763325901154f * dz * (5.f * z2 - 3.f));
      R[13] = (f16)( 0.4570457994644657f * dx * (1.f - 5.f * z2));
      R[14] = (f16)( 1.445305721320277f * dz * (x2 - y2));
      R[15] = (f16)( 0.5900435899266435f * dx * (-x2 + 3.f * y2));
    }
  }

  __syncthreads();  // only barrier: after this every wave owns its 16 rows

  // ---- WMMA MLP: one 16-point row tile per wave ----
  const int lane = tid & 31;
  const int row0 = (tid >> 5) * 16;
  const int col  = lane & 15;
  const int rb   = (lane & 16) ? 8 : 0;
  const v8f vzero = {};

  // sigma layer 1: relu(enc[16x32] @ ws1[32x64]) -> sH
  {
    const v16h a = ldsAfrag(sA, 40, row0, 0, lane);
#pragma unroll
    for (int nt = 0; nt < 4; ++nt) {
      const v8f acc = wmma16(a, ldsBfrag(sWs1, 32, nt * 16, 0, lane), vzero);
      storeCrelu(sH, 72, row0, nt * 16, lane, acc);
    }
  }

  // geometry: sH[16x64] @ ws2[64x32pad]; col0 -> sigma, cols 1..16 -> geo_feat
  {
    const v16h h0 = ldsAfrag(sH, 72, row0, 0, lane);
    const v16h h1 = ldsAfrag(sH, 72, row0, 1, lane);
    v8f g0 = wmma16(h0, ldsBfrag(sWs2, 64, 0, 0, lane), vzero);
    g0     = wmma16(h1, ldsBfrag(sWs2, 64, 0, 1, lane), g0);
    v8f g1 = wmma16(h0, ldsBfrag(sWs2, 64, 16, 0, lane), vzero);
    g1     = wmma16(h1, ldsBfrag(sWs2, 64, 16, 1, lane), g1);
#pragma unroll
    for (int r = 0; r < 8; ++r) {
      const int prow = row0 + rb + r;
      const int gpt  = base + prow;
      if (col == 0) {
        if (gpt < N) sigma_out[gpt] = fmaxf(g0[r], 0.0f);   // relu(sigma)
        sR[prow * 40 + 31] = (f16)g1[r];                    // geometry col 16
      } else {
        sR[prow * 40 + 15 + col] = (f16)g0[r];              // geometry cols 1..15
      }
    }
  }

  // rgb layer 1: relu(rgb_in[16x32] @ wr1[32x64]) -> sH
  {
    const v16h a = ldsAfrag(sR, 40, row0, 0, lane);
#pragma unroll
    for (int nt = 0; nt < 4; ++nt) {
      const v8f acc = wmma16(a, ldsBfrag(sWr1, 32, nt * 16, 0, lane), vzero);
      storeCrelu(sH, 72, row0, nt * 16, lane, acc);
    }
  }

  // rgb layer 2: relu(sH[16x64] @ wr2[64x64]) -> sH (frags read before stores)
  {
    const v16h h0 = ldsAfrag(sH, 72, row0, 0, lane);
    const v16h h1 = ldsAfrag(sH, 72, row0, 1, lane);
    v8f acc[4];
#pragma unroll
    for (int nt = 0; nt < 4; ++nt) {
      acc[nt] = wmma16(h0, ldsBfrag(sWr2, 64, nt * 16, 0, lane), vzero);
      acc[nt] = wmma16(h1, ldsBfrag(sWr2, 64, nt * 16, 1, lane), acc[nt]);
    }
#pragma unroll
    for (int nt = 0; nt < 4; ++nt)
      storeCrelu(sH, 72, row0, nt * 16, lane, acc[nt]);
  }

  // rgb layer 3: sigmoid(sH[16x64] @ wr3[64x16pad]) cols 0..2 -> rgb_out
  {
    const v16h h0 = ldsAfrag(sH, 72, row0, 0, lane);
    const v16h h1 = ldsAfrag(sH, 72, row0, 1, lane);
    v8f o = wmma16(h0, ldsBfrag(sWr3, 64, 0, 0, lane), vzero);
    o     = wmma16(h1, ldsBfrag(sWr3, 64, 0, 1, lane), o);
    if (col < 3) {
#pragma unroll
      for (int r = 0; r < 8; ++r) {
        const int gpt = base + row0 + rb + r;
        if (gpt < N) rgb_out[gpt * 3 + col] = 1.0f / (1.0f + __expf(-o[r]));
      }
    }
  }
}

// ---------------- launch -----------------------------------------------------

extern "C" void kernel_launch(void* const* d_in, const int* in_sizes, int n_in,
                              void* d_out, int out_size, void* d_ws,
                              size_t ws_size, hipStream_t stream) {
  const float* xyz   = (const float*)d_in[0];
  const float* dir   = (const float*)d_in[1];
  const float* table = (const float*)d_in[2];
  const float* ws1   = (const float*)d_in[3];
  const float* ws2   = (const float*)d_in[4];
  const float* wr1   = (const float*)d_in[5];
  const float* wr2   = (const float*)d_in[6];
  const float* wr3   = (const float*)d_in[7];

  const int N = in_sizes[0] / 3;                 // 2^21 points
  float* rgb   = (float*)d_out;                  // N x 3, first in tuple
  float* sigma = rgb + (size_t)3 * N;            // N x 1, second in tuple

  const int blocks = (N + 127) / 128;            // 128 points / block (8 waves)
  nerf_fused<<<blocks, 256, 0, stream>>>(xyz, dir, table, ws1, ws2, wr1, wr2,
                                         wr3, rgb, sigma, N);
}
